// EdgeConvNet_54039278519135
// MI455X (gfx1250) — matrix operand order repaired and checked by reading, not compiled, not run on hardware
//
#include <hip/hip_runtime.h>
#include <hip/hip_bf16.h>
#include <math.h>

typedef __attribute__((ext_vector_type(16))) _Float16 v16h;
typedef __attribute__((ext_vector_type(8)))  _Float16 v8h;
typedef __attribute__((ext_vector_type(8)))  float    v8f;

#define WAVES 8
#define EPSBN 1e-5f

static __device__ __forceinline__ v8f wmma_f16(v16h a, v16h b, v8f c) {
  return __builtin_amdgcn_wmma_f32_16x16x32_f16(false, a, false, b, (short)0, c, false, false);
}

static __device__ __forceinline__ v16h combine16(v8h lo, v8h hi) {
  v16h r;
#pragma unroll
  for (int i = 0; i < 8; ++i) { r[i] = lo[i]; r[8 + i] = hi[i]; }
  return r;
}

// ---------------------------------------------------------------------------
// K0: fold BatchNorm into weights, emit PRE-SWIZZLED into WMMA B-fragment
//     order: dst[blk*512 + lane*16 + i] =
//        W[(kc*32 + (lane>>4)*16 + i)*NC + nt*16 + (lane&15)],  blk = kc*NT+nt
// ---------------------------------------------------------------------------
__global__ void prep_kernel(
    const float* __restrict__ w0, const float* __restrict__ b0, const float* __restrict__ g0,
    const float* __restrict__ be0, const float* __restrict__ m0, const float* __restrict__ v0,
    const float* __restrict__ w1, const float* __restrict__ b1, const float* __restrict__ g1,
    const float* __restrict__ be1, const float* __restrict__ m1, const float* __restrict__ v1,
    const float* __restrict__ w2, const float* __restrict__ b2, const float* __restrict__ g2,
    const float* __restrict__ be2, const float* __restrict__ m2, const float* __restrict__ v2,
    const float* __restrict__ we0, const float* __restrict__ bee0,
    const float* __restrict__ we1, const float* __restrict__ bee1,
    _Float16* __restrict__ W0H, _Float16* __restrict__ W1H, _Float16* __restrict__ W2H,
    float* __restrict__ B0, float* __restrict__ B1, float* __restrict__ B2,
    _Float16* __restrict__ WE0H, float* __restrict__ BEE0,
    float* __restrict__ WE1, float* __restrict__ BEE1)
{
  int tid = blockIdx.x * blockDim.x + threadIdx.x;
  int nth = gridDim.x * blockDim.x;

  // W0: K=32, NC=64 (NT=4, 4 blocks, 2048 elems)
  for (int d = tid; d < 2048; d += nth) {
    int i = d & 15, l = (d >> 4) & 31, b = d >> 9;
    int nt = b & 3, kc = b >> 2;
    int row = kc * 32 + ((l >> 4) << 4) + i;
    int col = (nt << 4) + (l & 15);
    float s = g0[col] * rsqrtf(v0[col] + EPSBN);
    W0H[d] = (_Float16)(w0[row * 64 + col] * s);
  }
  // W1/W2: K=64, NC=64 (NT=4, 8 blocks, 4096 elems)
  for (int d = tid; d < 4096; d += nth) {
    int i = d & 15, l = (d >> 4) & 31, b = d >> 9;
    int nt = b & 3, kc = b >> 2;
    int row = kc * 32 + ((l >> 4) << 4) + i;
    int col = (nt << 4) + (l & 15);
    float s1 = g1[col] * rsqrtf(v1[col] + EPSBN);
    float s2 = g2[col] * rsqrtf(v2[col] + EPSBN);
    W1H[d] = (_Float16)(w1[row * 64 + col] * s1);
    W2H[d] = (_Float16)(w2[row * 64 + col] * s2);
  }
  // biases
  for (int n = tid; n < 64; n += nth) {
    float s0 = g0[n] * rsqrtf(v0[n] + EPSBN); B0[n] = (b0[n] - m0[n]) * s0 + be0[n];
    float s1 = g1[n] * rsqrtf(v1[n] + EPSBN); B1[n] = (b1[n] - m1[n]) * s1 + be1[n];
    float s2 = g2[n] * rsqrtf(v2[n] + EPSBN); B2[n] = (b2[n] - m2[n]) * s2 + be2[n];
  }
  // WE0: K=96 (padded from 80), NC=128 (NT=8, 24 blocks, 12288 elems)
  for (int d = tid; d < 12288; d += nth) {
    int i = d & 15, l = (d >> 4) & 31, b = d >> 9;
    int nt = b & 7, kc = b >> 3;
    int row = kc * 32 + ((l >> 4) << 4) + i;
    int col = (nt << 4) + (l & 15);
    WE0H[d] = (_Float16)((row < 80) ? we0[row * 128 + col] : 0.0f);
  }
  for (int i = tid; i < 128; i += nth) { BEE0[i] = bee0[i]; WE1[i] = we1[i]; }
  if (tid == 0) BEE1[0] = bee1[0];
}

// ---------------------------------------------------------------------------
// K1: zero segment accumulators (harness does not reset between replays)
// ---------------------------------------------------------------------------
__global__ void zero_ws_kernel(float* __restrict__ seg, float* __restrict__ cnt, int N)
{
  int tid = blockIdx.x * blockDim.x + threadIdx.x;
  int nth = gridDim.x * blockDim.x;
  int total = N * 64;
  for (int i = tid; i < total; i += nth) seg[i] = 0.0f;
  for (int i = tid; i < N; i += nth) cnt[i] = 0.0f;
}

// ---------------------------------------------------------------------------
// Fragment loads (all 16B-vector LDS accesses)
// ---------------------------------------------------------------------------
static __device__ __forceinline__ v16h load_a_frag(const _Float16* __restrict__ rowbase,
                                                   int stride, int k0, int lane)
{
  int hi = (lane & 16) ? 16 : 0;
  const _Float16* p = rowbase + (lane & 15) * stride + k0 + hi;
  return combine16(*(const v8h*)p, *(const v8h*)(p + 8));
}

static __device__ __forceinline__ v16h load_b_sw(const _Float16* __restrict__ sW,
                                                 int blk, int lane)
{
  const _Float16* p = sW + blk * 512 + lane * 16;
  return combine16(*(const v8h*)p, *(const v8h*)(p + 8));
}

// out[16][64] = relu(in[16][K] @ W + bias); A fragments hoisted across N-tiles
template <int KCH>
static __device__ __forceinline__ void mlp_layer(const _Float16* __restrict__ sIn,
                                                 const _Float16* __restrict__ sW,
                                                 const float* __restrict__ sB,
                                                 _Float16* __restrict__ sOut, int lane)
{
  int nlo = lane & 15;
  int mb = (lane & 16) ? 8 : 0;
  v16h a[KCH];
#pragma unroll
  for (int kc = 0; kc < KCH; ++kc) a[kc] = load_a_frag(sIn, 64, kc * 32, lane);
#pragma unroll
  for (int nt = 0; nt < 4; ++nt) {
    int n = nt * 16 + nlo;
    v8f c;
    float bias = sB[n];
#pragma unroll
    for (int r = 0; r < 8; ++r) c[r] = bias;
#pragma unroll
    for (int kc = 0; kc < KCH; ++kc)
      c = wmma_f16(a[kc], load_b_sw(sW, kc * 4 + nt, lane), c);
#pragma unroll
    for (int r = 0; r < 8; ++r)
      sOut[(mb + r) * 64 + n] = (_Float16)fmaxf(c[r], 0.0f);
  }
}

// Last layer: relu output scattered via atomics into seg[node][64]
static __device__ __forceinline__ void mlp_layer_scatter(const _Float16* __restrict__ sIn,
                                                         const _Float16* __restrict__ sW,
                                                         const float* __restrict__ sB,
                                                         const int* __restrict__ sDstRow,
                                                         float* __restrict__ seg, int lane)
{
  int nlo = lane & 15;
  int mb = (lane & 16) ? 8 : 0;
  v16h a[2];
#pragma unroll
  for (int kc = 0; kc < 2; ++kc) a[kc] = load_a_frag(sIn, 64, kc * 32, lane);
#pragma unroll
  for (int nt = 0; nt < 4; ++nt) {
    int n = nt * 16 + nlo;
    v8f c;
    float bias = sB[n];
#pragma unroll
    for (int r = 0; r < 8; ++r) c[r] = bias;
#pragma unroll
    for (int kc = 0; kc < 2; ++kc)
      c = wmma_f16(a[kc], load_b_sw(sW, kc * 4 + nt, lane), c);
#pragma unroll
    for (int r = 0; r < 8; ++r) {
      int node = sDstRow[mb + r];
      if (node >= 0)
        atomicAdd(seg + (size_t)node * 64 + n, fmaxf(c[r], 0.0f));
    }
  }
}

// ---------------------------------------------------------------------------
// K2: per-edge 3-layer MLP + atomic segment-sum.  16 edges/wave, 8 waves/block
// ---------------------------------------------------------------------------
__global__ __launch_bounds__(256) void edge_mlp_kernel(
    const float* __restrict__ x, const int* __restrict__ ei,
    const _Float16* __restrict__ W0H, const _Float16* __restrict__ W1H,
    const _Float16* __restrict__ W2H,
    const float* __restrict__ B0, const float* __restrict__ B1, const float* __restrict__ B2,
    float* __restrict__ seg, float* __restrict__ cnt, int E)
{
  __shared__ __align__(16) _Float16 sW0[32 * 64];
  __shared__ __align__(16) _Float16 sW1[64 * 64];
  __shared__ __align__(16) _Float16 sW2[64 * 64];
  __shared__ float sB0[64], sB1[64], sB2[64];
  __shared__ __align__(16) _Float16 sHa[WAVES][16 * 64];
  __shared__ __align__(16) _Float16 sHb[WAVES][16 * 64];
  __shared__ int sDst[WAVES][16];

  int t = threadIdx.x;
  {
    v8h* d0 = (v8h*)sW0; const v8h* s0 = (const v8h*)W0H;
    for (int i = t; i < 2048 / 8; i += 256) d0[i] = s0[i];
    v8h* d1 = (v8h*)sW1; const v8h* s1 = (const v8h*)W1H;
    for (int i = t; i < 4096 / 8; i += 256) d1[i] = s1[i];
    v8h* d2 = (v8h*)sW2; const v8h* s2 = (const v8h*)W2H;
    for (int i = t; i < 4096 / 8; i += 256) d2[i] = s2[i];
    if (t < 64) { sB0[t] = B0[t]; sB1[t] = B1[t]; sB2[t] = B2[t]; }
  }

  int w = t >> 5, lane = t & 31;
  int eBase = blockIdx.x * (WAVES * 16) + w * 16;

  // Stage inputs: h = [x_i, x_j - x_i] as f16, 32 halves/row (vector stores)
  if (lane < 16) {
    int e = eBase + lane;
    int src = 0, dst = 0;
    bool valid = (e < E);
    if (valid) { src = ei[e]; dst = ei[E + e]; }
    sDst[w][lane] = valid ? dst : -1;
    const float4* xi4 = (const float4*)(x + (size_t)dst * 16);
    const float4* xj4 = (const float4*)(x + (size_t)src * 16);
    _Float16 hv[32];
#pragma unroll
    for (int q = 0; q < 4; ++q) {
      float4 a = xi4[q], bq = xj4[q];
      hv[q * 4 + 0]      = (_Float16)a.x;
      hv[q * 4 + 1]      = (_Float16)a.y;
      hv[q * 4 + 2]      = (_Float16)a.z;
      hv[q * 4 + 3]      = (_Float16)a.w;
      hv[16 + q * 4 + 0] = (_Float16)(bq.x - a.x);
      hv[16 + q * 4 + 1] = (_Float16)(bq.y - a.y);
      hv[16 + q * 4 + 2] = (_Float16)(bq.z - a.z);
      hv[16 + q * 4 + 3] = (_Float16)(bq.w - a.w);
    }
    v8h* dstp = (v8h*)&sHa[w][lane * 64];
#pragma unroll
    for (int c = 0; c < 4; ++c) {
      v8h tv;
#pragma unroll
      for (int i = 0; i < 8; ++i) tv[i] = hv[c * 8 + i];
      dstp[c] = tv;
    }
    if (valid) atomicAdd(&cnt[dst], 1.0f);
  }
  __syncthreads();

  mlp_layer<1>(sHa[w], sW0, sB0, sHb[w], lane);             // 32 -> 64
  __syncthreads();
  mlp_layer<2>(sHb[w], sW1, sB1, sHa[w], lane);             // 64 -> 64
  __syncthreads();
  mlp_layer_scatter(sHa[w], sW2, sB2, sDst[w], seg, lane);  // 64 -> 64 + scatter
}

// ---------------------------------------------------------------------------
// K3: y = relu([seg/max(cnt,1), x]) stored f16 padded to 96 features
// ---------------------------------------------------------------------------
__global__ void node_y_kernel(const float* __restrict__ x, const float* __restrict__ seg,
                              const float* __restrict__ cnt, _Float16* __restrict__ yh, int N)
{
  int i = blockIdx.x * blockDim.x + threadIdx.x;
  if (i >= N * 96) return;
  int node = i / 96, f = i - node * 96;
  float v;
  if (f < 64) {
    float c = fmaxf(cnt[node], 1.0f);
    v = seg[(size_t)node * 64 + f] / c;
  } else if (f < 80) {
    v = x[(size_t)node * 16 + (f - 64)];
  } else {
    v = 0.0f;
  }
  yh[i] = (_Float16)fmaxf(v, 0.0f);
}

// ---------------------------------------------------------------------------
// K4: e = y[src]-y[dst]; z = relu(e@we0+bee0); out = sigmoid(z@we1+bee1)
//     A fragments hoisted across the 8 N-tiles; fused 128->1 dot via
//     per-lane accumulation + shfl-xor reduction within 16-lane groups.
// ---------------------------------------------------------------------------
__global__ __launch_bounds__(256) void edge_out_kernel(
    const _Float16* __restrict__ yh, const int* __restrict__ ei,
    const _Float16* __restrict__ WE0H, const float* __restrict__ BEE0,
    const float* __restrict__ WE1, const float* __restrict__ BEE1,
    float* __restrict__ out, int E)
{
  __shared__ __align__(16) _Float16 sWE0[96 * 128];
  __shared__ float sBE0[128], sWE1[128];
  __shared__ __align__(16) _Float16 sE[WAVES][16 * 96];

  int t = threadIdx.x;
  {
    v8h* d0 = (v8h*)sWE0; const v8h* s0 = (const v8h*)WE0H;
    for (int i = t; i < 12288 / 8; i += 256) d0[i] = s0[i];
    if (t < 128) { sBE0[t] = BEE0[t]; sWE1[t] = WE1[t]; }
  }

  int w = t >> 5, lane = t & 31;
  int eBase = blockIdx.x * (WAVES * 16) + w * 16;

  if (lane < 16) {
    int e = eBase + lane;
    int src = 0, dst = 0;
    if (e < E) { src = ei[e]; dst = ei[E + e]; }
    const v8h* ys8 = (const v8h*)(yh + (size_t)src * 96);
    const v8h* yd8 = (const v8h*)(yh + (size_t)dst * 96);
    v8h* se8 = (v8h*)&sE[w][lane * 96];
#pragma unroll
    for (int c = 0; c < 12; ++c) se8[c] = ys8[c] - yd8[c];  // packed f16 sub
  }
  __syncthreads();

  int nlo = lane & 15;
  v16h a[3];
#pragma unroll
  for (int kc = 0; kc < 3; ++kc) a[kc] = load_a_frag(sE[w], 96, kc * 32, lane);

  float zacc[8];
#pragma unroll
  for (int r = 0; r < 8; ++r) zacc[r] = 0.0f;

#pragma unroll
  for (int nt = 0; nt < 8; ++nt) {
    int n = nt * 16 + nlo;
    v8f c;
    float bias = sBE0[n];
#pragma unroll
    for (int r = 0; r < 8; ++r) c[r] = bias;
#pragma unroll
    for (int kc = 0; kc < 3; ++kc)
      c = wmma_f16(a[kc], load_b_sw(sWE0, kc * 8 + nt, lane), c);
    float wn = sWE1[n];
#pragma unroll
    for (int r = 0; r < 8; ++r) zacc[r] += fmaxf(c[r], 0.0f) * wn;
  }

  // Reduce across each 16-lane group (xor 8,4,2,1 stays within the group)
#pragma unroll
  for (int s = 8; s >= 1; s >>= 1) {
#pragma unroll
    for (int r = 0; r < 8; ++r) zacc[r] += __shfl_xor(zacc[r], s, 32);
  }

  if (nlo == 0) {
    int mb = (lane & 16) ? 8 : 0;
    float bee1v = BEE1[0];
#pragma unroll
    for (int r = 0; r < 8; ++r) {
      int e = eBase + mb + r;
      if (e < E) out[e] = 1.0f / (1.0f + __expf(-(zacc[r] + bee1v)));
    }
  }
}

// ---------------------------------------------------------------------------
// Launch
// ---------------------------------------------------------------------------
extern "C" void kernel_launch(void* const* d_in, const int* in_sizes, int n_in,
                              void* d_out, int out_size, void* d_ws, size_t ws_size,
                              hipStream_t stream)
{
  (void)n_in; (void)out_size; (void)ws_size;

  const float* x   = (const float*)d_in[0];
  const int*   ei  = (const int*)d_in[1];
  const float* w0  = (const float*)d_in[2];
  const float* b0  = (const float*)d_in[3];
  const float* g0  = (const float*)d_in[4];
  const float* be0 = (const float*)d_in[5];
  const float* m0  = (const float*)d_in[6];
  const float* v0  = (const float*)d_in[7];
  const float* w1  = (const float*)d_in[8];
  const float* b1  = (const float*)d_in[9];
  const float* g1  = (const float*)d_in[10];
  const float* be1 = (const float*)d_in[11];
  const float* m1  = (const float*)d_in[12];
  const float* v1  = (const float*)d_in[13];
  const float* w2  = (const float*)d_in[14];
  const float* b2  = (const float*)d_in[15];
  const float* g2  = (const float*)d_in[16];
  const float* be2 = (const float*)d_in[17];
  const float* m2  = (const float*)d_in[18];
  const float* v2  = (const float*)d_in[19];
  const float* we0 = (const float*)d_in[20];
  const float* bee0= (const float*)d_in[21];
  const float* we1 = (const float*)d_in[22];
  const float* bee1= (const float*)d_in[23];

  const int N = in_sizes[0] / 16;   // 100000
  const int E = in_sizes[1] / 2;    // 1600000

  // Workspace layout (byte offsets, 256-aligned)
  char* ws = (char*)d_ws;
  _Float16* W0H  = (_Float16*)(ws + 0);        // 2048  f16 (swizzled)
  _Float16* W1H  = (_Float16*)(ws + 4096);     // 4096  f16 (swizzled)
  _Float16* W2H  = (_Float16*)(ws + 12288);    // 4096  f16 (swizzled)
  float*    B0   = (float*)   (ws + 20480);    // 64 f32
  float*    B1   = (float*)   (ws + 20736);
  float*    B2   = (float*)   (ws + 20992);
  _Float16* WE0H = (_Float16*)(ws + 21504);    // 12288 f16 (swizzled, K padded 96)
  float*    BEE0 = (float*)   (ws + 46080);    // 128 f32
  float*    WE1  = (float*)   (ws + 46592);    // 128 f32
  float*    BEE1 = (float*)   (ws + 47104);    // 1 f32
  float*    seg  = (float*)   (ws + 47360);                      // N*64 f32
  float*    cnt  = (float*)   (ws + 47360 + (size_t)N * 64 * 4); // N f32
  _Float16* yh   = (_Float16*)(ws + 47360 + (size_t)N * 65 * 4); // N*96 f16

  prep_kernel<<<48, 256, 0, stream>>>(
      w0, b0, g0, be0, m0, v0, w1, b1, g1, be1, m1, v1,
      w2, b2, g2, be2, m2, v2, we0, bee0, we1, bee1,
      W0H, W1H, W2H, B0, B1, B2, WE0H, BEE0, WE1, BEE1);

  {
    int total = N * 64;
    int blocks = (total + 255) / 256;
    if (blocks > 4096) blocks = 4096;
    zero_ws_kernel<<<blocks, 256, 0, stream>>>(seg, cnt, N);
  }

  {
    int blocks = (E + WAVES * 16 - 1) / (WAVES * 16);
    edge_mlp_kernel<<<blocks, 256, 0, stream>>>(x, ei, W0H, W1H, W2H,
                                                B0, B1, B2, seg, cnt, E);
  }

  {
    int blocks = (N * 96 + 255) / 256;
    node_y_kernel<<<blocks, 256, 0, stream>>>(x, seg, cnt, yh, N);
  }

  {
    int blocks = (E + WAVES * 16 - 1) / (WAVES * 16);
    edge_out_kernel<<<blocks, 256, 0, stream>>>(yh, ei, WE0H, BEE0, WE1, BEE1,
                                                (float*)d_out, E);
  }
}
